// SparseAttention_36670430773877
// MI455X (gfx1250) — matrix operand order repaired and checked by reading, not compiled, not run on hardware
//
#include <hip/hip_runtime.h>
#include <stdint.h>

// ---------------------------------------------------------------------------
// Types
// ---------------------------------------------------------------------------
typedef __bf16 bf16;
typedef __bf16 v16bf __attribute__((ext_vector_type(16)));
typedef float  v8f   __attribute__((ext_vector_type(8)));
typedef uint32_t u32;
typedef uint32_t u32x4 __attribute__((ext_vector_type(4)));
typedef int32_t  i32x4 __attribute__((ext_vector_type(4)));
typedef int32_t  i32x8 __attribute__((ext_vector_type(8)));

#define S_LEN  4096
#define DMODEL 1024
#define NHEAD  16
#define HDIM   64
#define BLK    128
#define NBLK   (S_LEN / BLK)      // 32
#define FIX2   33                 // columns per earlier block (1 + 32)
#define NEGF   (-1.0e30f)

// ---------------------------------------------------------------------------
// Small helpers
// ---------------------------------------------------------------------------
__device__ inline v8f v8zero() {
  v8f z;
#pragma unroll
  for (int e = 0; e < 8; ++e) z[e] = 0.0f;
  return z;
}

__device__ inline v8f wmma_bf16(v16bf a, v16bf b, v8f c) {
  // 8 args: (neg_a, A, neg_b, B, c_mod, C, reuse_a, reuse_b)
  return __builtin_amdgcn_wmma_f32_16x16x32_bf16(false, a, false, b,
                                                 (short)0, c, false, false);
}

union FragU { v16bf v; u32 u[8]; };

// A fragment 16x32 bf16 from row-major [row][k] storage (LDS).
// Layout: lanes 0-15 hold row M, K=0..15;
// lanes 16-31 hold row M, K=16..31 (matches load_b_frag).
__device__ inline v16bf load_a_frag(const bf16* base, int stride, int row0, int k0) {
  const int lane = threadIdx.x & 31;
  const int l = lane & 15, hi = lane >> 4;
  const u32* rp = (const u32*)(base + (row0 + l) * stride + k0);
  FragU f;
#pragma unroll
  for (int j = 0; j < 8; ++j) {
    const int kp = j + 8 * hi;   // uint (K-pair) index
    f.u[j] = rp[kp];
  }
  return f.v;
}

// B fragment 32x16 bf16 from [n][k] storage (LDS).
// lanes 0-15: col N=l, K=0..15; lanes 16-31: col N=l, K=16..31.
__device__ inline v16bf load_b_frag(const bf16* base, int stride, int n0, int k0) {
  const int lane = threadIdx.x & 31;
  const int l = lane & 15, hi = lane >> 4;
  const u32* rp = (const u32*)(base + (n0 + l) * stride + k0 + 16 * hi);
  FragU f;
#pragma unroll
  for (int j = 0; j < 8; ++j) f.u[j] = rp[j];
  return f.v;
}

__device__ inline float half16_max(float v) {
#pragma unroll
  for (int m = 1; m < 16; m <<= 1) v = fmaxf(v, __shfl_xor(v, m, 16));
  return v;
}
__device__ inline float half16_sum(float v) {
#pragma unroll
  for (int m = 1; m < 16; m <<= 1) v += __shfl_xor(v, m, 16);
  return v;
}

// ---------------------------------------------------------------------------
// Tensor Data Mover: 2-D bf16 tile (tile_h rows x tile_w elems) -> LDS.
// D# fields per CDNA5 ISA. One wave issues it.
// ---------------------------------------------------------------------------
__device__ inline void tdm_load_2d_bf16(u32 lds_off_bytes, const bf16* gptr,
                                        u32 tile_w, u32 tile_h,
                                        u32 row_stride_elems) {
  const unsigned long long ga = (unsigned long long)(uintptr_t)gptr;
  u32x4 g0;
  g0[0] = 1u;                                              // count=1, user mode
  g0[1] = lds_off_bytes;                                   // lds_addr
  g0[2] = (u32)(ga & 0xffffffffull);                       // global_addr lo
  g0[3] = (u32)((ga >> 32) & 0x1ffffffull) | (2u << 30);   // global hi | type=2
  i32x8 g1;
  g1[0] = (int)(1u << 16);                                 // data_size=1 (2 B)
  g1[1] = (int)((tile_w & 0xffffu) << 16);                 // tensor_dim0 lo16
  g1[2] = (int)((tile_w >> 16) | ((tile_h & 0xffffu) << 16)); // dim0 hi|dim1 lo
  g1[3] = (int)((tile_h >> 16) | (tile_w << 16));          // dim1 hi | tile_dim0
  g1[4] = (int)(tile_h & 0xffffu);                         // tile_dim1, tile_dim2=0
  g1[5] = (int)row_stride_elems;                           // dim0 stride lo32
  g1[6] = 0;
  g1[7] = 0;
  i32x4 z4 = {0, 0, 0, 0};
#if __has_include(<hip/amd_detail/amd_gfx1250_TDM.h>)
  i32x8 z8 = {0, 0, 0, 0, 0, 0, 0, 0};
  __builtin_amdgcn_tensor_load_to_lds(g0, g1, z4, z4, z8, 0);
#else
  __builtin_amdgcn_tensor_load_to_lds(g0, g1, z4, z4, 0);
#endif
}

// ---------------------------------------------------------------------------
// Kernel 1: fp32 -> bf16 conversion; weights transposed to [N][K].
// ---------------------------------------------------------------------------
__global__ __launch_bounds__(256) void convert_kernel(
    const float* __restrict__ x,  const float* __restrict__ wq,
    const float* __restrict__ wk, const float* __restrict__ wv,
    const float* __restrict__ wo,
    bf16* __restrict__ xb, bf16* __restrict__ wqt, bf16* __restrict__ wkt,
    bf16* __restrict__ wvt, bf16* __restrict__ wot) {
  const int stride = gridDim.x * blockDim.x;
  const int t0 = blockIdx.x * blockDim.x + threadIdx.x;
  for (int i = t0; i < S_LEN * DMODEL; i += stride) xb[i] = (bf16)x[i];
  for (int i = t0; i < DMODEL * DMODEL; i += stride) {
    const int r = i >> 10, c = i & 1023;        // input [r][c] -> out [c][r]
    wqt[c * DMODEL + r] = (bf16)wq[i];
    wkt[c * DMODEL + r] = (bf16)wk[i];
    wvt[c * DMODEL + r] = (bf16)wv[i];
    wot[c * DMODEL + r] = (bf16)wo[i];
  }
}

// ---------------------------------------------------------------------------
// Kernel 2/4: C[4096x1024] = A[4096x1024] * Bt[1024x1024]^T ; out = (C+bias)*scale
// A row-major [M][K] bf16, Bt row-major [N][K] bf16.
// Block = 128x128 tile, 8 waves, each wave 2(M) x 4(N) WMMA accumulators.
// ---------------------------------------------------------------------------
__global__ __launch_bounds__(256) void gemm_bf16(
    const bf16* __restrict__ A, const bf16* __restrict__ Bt,
    const float* __restrict__ bias, float scale,
    bf16* __restrict__ outb, float* __restrict__ outf) {
  constexpr int Kdim = DMODEL, Ndim = DMODEL, LDT = 40;  // padded LDS stride
  __shared__ __attribute__((aligned(16))) bf16 Al[128 * LDT];
  __shared__ __attribute__((aligned(16))) bf16 Bl[128 * LDT];

  const int mb = blockIdx.x, nb = blockIdx.y;
  const int tid = threadIdx.x, lane = tid & 31, w = tid >> 5;
  const int wm = w >> 1, wn = w & 1;
  const int l = lane & 15, hi = lane >> 4;

  v8f acc[2][4];
#pragma unroll
  for (int mt = 0; mt < 2; ++mt)
#pragma unroll
    for (int nt = 0; nt < 4; ++nt) acc[mt][nt] = v8zero();

  const int row = tid >> 1, seg = tid & 1;   // each thread stages 16 bf16/row-half
  const bf16* Ag = A  + (size_t)(mb * 128 + row) * Kdim + seg * 16;
  const bf16* Bg = Bt + (size_t)(nb * 128 + row) * Kdim + seg * 16;
  bf16* Als = Al + row * LDT + seg * 16;
  bf16* Bls = Bl + row * LDT + seg * 16;

  for (int kk = 0; kk < Kdim; kk += 32) {
    const uint4 a0 = ((const uint4*)(Ag + kk))[0];
    const uint4 a1 = ((const uint4*)(Ag + kk))[1];
    const uint4 b0 = ((const uint4*)(Bg + kk))[0];
    const uint4 b1 = ((const uint4*)(Bg + kk))[1];
    __builtin_prefetch(Ag + kk + 64, 0, 0);   // global_prefetch_b8, K tile two steps ahead
    __builtin_prefetch(Bg + kk + 64, 0, 0);
    ((uint4*)Als)[0] = a0; ((uint4*)Als)[1] = a1;
    ((uint4*)Bls)[0] = b0; ((uint4*)Bls)[1] = b1;
    __syncthreads();

    v16bf af[2];
#pragma unroll
    for (int mt = 0; mt < 2; ++mt)
      af[mt] = load_a_frag(Al, LDT, wm * 32 + mt * 16, 0);
#pragma unroll
    for (int nt = 0; nt < 4; ++nt) {
      const v16bf bf_ = load_b_frag(Bl, LDT, wn * 64 + nt * 16, 0);
#pragma unroll
      for (int mt = 0; mt < 2; ++mt)
        acc[mt][nt] = wmma_bf16(af[mt], bf_, acc[mt][nt]);
    }
    __syncthreads();
  }

#pragma unroll
  for (int mt = 0; mt < 2; ++mt)
#pragma unroll
    for (int nt = 0; nt < 4; ++nt)
#pragma unroll
      for (int i = 0; i < 8; ++i) {
        const int gm = mb * 128 + wm * 32 + mt * 16 + i + 8 * hi;
        const int gn = nb * 128 + wn * 64 + nt * 16 + l;
        const float v = (acc[mt][nt][i] + bias[gn]) * scale;
        if (outb) outb[(size_t)gm * Ndim + gn] = (bf16)v;
        else      outf[(size_t)gm * Ndim + gn] = v;
      }
}

// ---------------------------------------------------------------------------
// Kernel 3: sparse flash attention. grid = (head, query-block), 256 threads.
// Chunks: G gathered fixed2 chunks (128 compact keys each) + causal diagonal.
// ---------------------------------------------------------------------------
__global__ __launch_bounds__(256) void attn_kernel(
    const bf16* __restrict__ Q, const bf16* __restrict__ Kg,
    const bf16* __restrict__ Vg, bf16* __restrict__ Ctx) {
  // LDS: [0,16384) Q-then-K tile 128x64 (stride 64)
  //      [16384,33792) V^T tile 64x136
  //      [33792,52224) P staging, 8 waves x 16 x 72
  __shared__ __attribute__((aligned(16))) unsigned char smem[52224];
  bf16* KQ = (bf16*)smem;
  bf16* VT = (bf16*)(smem + 16384);
  bf16* Pl = (bf16*)(smem + 33792);

  const int h  = blockIdx.x;
  const int qb = blockIdx.y;
  const int tid = threadIdx.x, lane = tid & 31, w = tid >> 5;
  const int l = lane & 15, hi = lane >> 4;

  // ---- stage Q tile via TDM (wave0 issues, waits TENSORcnt, then WG barrier)
  if (tid < 32) {
    tdm_load_2d_bf16(0u, Q + (size_t)(qb * BLK) * DMODEL + h * HDIM,
                     HDIM, BLK, DMODEL);
    __builtin_amdgcn_s_wait_tensorcnt(0);
  }
  __syncthreads();
  v16bf qf[2];
#pragma unroll
  for (int ks = 0; ks < 2; ++ks) qf[ks] = load_a_frag(KQ, HDIM, w * 16, ks * 32);
  __syncthreads();   // Q region will be reused for K tiles

  v8f acc[4];
#pragma unroll
  for (int t = 0; t < 4; ++t) acc[t] = v8zero();
  float mrun[8], lrun[8];
#pragma unroll
  for (int i = 0; i < 8; ++i) { mrun[i] = -3.0e38f; lrun[i] = 0.0f; }

  const int nsp = FIX2 * qb;            // gathered compact keys
  const int G = (nsp + 127) >> 7;       // gather chunks
  bf16* Pw = Pl + w * 16 * 72;

  for (int c = 0; c <= G; ++c) {
    const bool diag = (c == G);
    int nvalid = 128;
    if (!diag) { const int rem = nsp - c * 128; nvalid = rem < 128 ? rem : 128; }

    // ---- stage K (TDM for diagonal, gathered rows otherwise) and V^T ----
    {
      const int j = tid >> 1, hf = tid & 1;   // key row j, 32-elem half hf
      int gk = 0; bool okrow = true;
      if (diag) {
        gk = qb * BLK + j;
      } else {
        const int jj = c * 128 + j;
        okrow = jj < nsp;
        if (okrow) {
          const int kb2 = jj / FIX2, r = jj - kb2 * FIX2;
          gk = kb2 * BLK + (r == 0 ? 0 : 95 + r);
        }
      }
      if (diag) {
        if (tid < 32) {
          tdm_load_2d_bf16(0u, Kg + (size_t)(qb * BLK) * DMODEL + h * HDIM,
                           HDIM, BLK, DMODEL);
          __builtin_amdgcn_s_wait_tensorcnt(0);
        }
      } else {
        union { uint4 q4[4]; bf16 e[32]; } kb_;
        const uint4* kr = (const uint4*)(Kg + (size_t)gk * DMODEL + h * HDIM + hf * 32);
        if (okrow) {
#pragma unroll
          for (int t = 0; t < 4; ++t) kb_.q4[t] = kr[t];
        } else {
#pragma unroll
          for (int e = 0; e < 32; ++e) kb_.e[e] = (bf16)0.0f;
        }
        uint4* kd = (uint4*)(KQ + j * HDIM + hf * 32);
#pragma unroll
        for (int t = 0; t < 4; ++t) kd[t] = kb_.q4[t];
      }
      // V rows, transposed on store into VT[hd][key]
      union { uint4 q4[4]; bf16 e[32]; } vb_;
      const uint4* vr = (const uint4*)(Vg + (size_t)gk * DMODEL + h * HDIM + hf * 32);
      if (okrow) {
#pragma unroll
        for (int t = 0; t < 4; ++t) vb_.q4[t] = vr[t];
      } else {
#pragma unroll
        for (int e = 0; e < 32; ++e) vb_.e[e] = (bf16)0.0f;
      }
#pragma unroll
      for (int e = 0; e < 32; ++e) VT[(hf * 32 + e) * 136 + j] = vb_.e[e];
    }
    __syncthreads();

    // ---- scores: 8 key sub-tiles x 2 K-steps of WMMA ----
    v8f sc[8];
#pragma unroll
    for (int nt = 0; nt < 8; ++nt) {
      sc[nt] = v8zero();
#pragma unroll
      for (int ks = 0; ks < 2; ++ks) {
        const v16bf kf = load_b_frag(KQ, HDIM, nt * 16, ks * 32);
        sc[nt] = wmma_bf16(qf[ks], kf, sc[nt]);
      }
    }

    // ---- mask ----
#pragma unroll
    for (int nt = 0; nt < 8; ++nt)
#pragma unroll
      for (int i = 0; i < 8; ++i) {
        const int jloc = nt * 16 + l;
        const bool ok = diag ? (jloc <= w * 16 + i + 8 * hi) : (jloc < nvalid);
        sc[nt][i] = ok ? sc[nt][i] : NEGF;
      }

    // ---- online softmax (row stats per VGPR index, 16-lane reductions) ----
    float corr[8];
#pragma unroll
    for (int i = 0; i < 8; ++i) {
      float m = sc[0][i];
#pragma unroll
      for (int nt = 1; nt < 8; ++nt) m = fmaxf(m, sc[nt][i]);
      m = half16_max(m);
      const float mn = fmaxf(mrun[i], m);
      corr[i] = __expf(mrun[i] - mn);
      mrun[i] = mn;
    }
#pragma unroll
    for (int nt = 0; nt < 8; ++nt)
#pragma unroll
      for (int i = 0; i < 8; ++i) sc[nt][i] = __expf(sc[nt][i] - mrun[i]);
#pragma unroll
    for (int i = 0; i < 8; ++i) {
      float s = 0.0f;
#pragma unroll
      for (int nt = 0; nt < 8; ++nt) s += sc[nt][i];
      lrun[i] = lrun[i] * corr[i] + half16_sum(s);
    }
#pragma unroll
    for (int t = 0; t < 4; ++t)
#pragma unroll
      for (int i = 0; i < 8; ++i) acc[t][i] *= corr[i];

    // ---- P*V in two passes of 64 keys (keeps P staging small) ----
#pragma unroll
    for (int ps = 0; ps < 2; ++ps) {
#pragma unroll
      for (int n2 = 0; n2 < 4; ++n2)
#pragma unroll
        for (int i = 0; i < 8; ++i)
          Pw[(i + 8 * hi) * 72 + n2 * 16 + l] = (bf16)sc[ps * 4 + n2][i];
#pragma unroll
      for (int kc = 0; kc < 2; ++kc) {
        const v16bf pa = load_a_frag(Pw, 72, 0, kc * 32);
#pragma unroll
        for (int t = 0; t < 4; ++t) {
          const v16bf vf = load_b_frag(VT, 136, t * 16, ps * 64 + kc * 32);
          acc[t] = wmma_bf16(pa, vf, acc[t]);
        }
      }
    }
    __syncthreads();   // before next chunk overwrites K/VT
  }

  // ---- normalize and write ctx (bf16) ----
#pragma unroll
  for (int i = 0; i < 8; ++i) lrun[i] = 1.0f / lrun[i];
#pragma unroll
  for (int t = 0; t < 4; ++t)
#pragma unroll
    for (int i = 0; i < 8; ++i) {
      const int gm = qb * BLK + w * 16 + i + 8 * hi;
      const int gn = h * HDIM + t * 16 + l;
      Ctx[(size_t)gm * DMODEL + gn] = (bf16)(acc[t][i] * lrun[i]);
    }
}

// ---------------------------------------------------------------------------
// Host-side launch
// ---------------------------------------------------------------------------
extern "C" void kernel_launch(void* const* d_in, const int* in_sizes, int n_in,
                              void* d_out, int out_size, void* d_ws, size_t ws_size,
                              hipStream_t stream) {
  (void)in_sizes; (void)n_in; (void)out_size; (void)ws_size;
  const float* x  = (const float*)d_in[0];
  const float* Wq = (const float*)d_in[1];
  const float* bq = (const float*)d_in[2];
  const float* Wk = (const float*)d_in[3];
  const float* bk = (const float*)d_in[4];
  const float* Wv = (const float*)d_in[5];
  const float* bv = (const float*)d_in[6];
  const float* Wo = (const float*)d_in[7];
  const float* bo = (const float*)d_in[8];

  bf16* ws  = (bf16*)d_ws;
  bf16* xb  = ws;                       // 4 Mi elems
  bf16* wqt = xb  + (4u << 20);         // 1 Mi each
  bf16* wkt = wqt + (1u << 20);
  bf16* wvt = wkt + (1u << 20);
  bf16* wot = wvt + (1u << 20);
  bf16* qB  = wot + (1u << 20);         // 4 Mi each
  bf16* kB  = qB  + (4u << 20);
  bf16* vB  = kB  + (4u << 20);
  bf16* ctx = vB  + (4u << 20);

  convert_kernel<<<1024, 256, 0, stream>>>(x, Wq, Wk, Wv, Wo,
                                           xb, wqt, wkt, wvt, wot);
  const dim3 gg(S_LEN / 128, DMODEL / 128);
  gemm_bf16<<<gg, 256, 0, stream>>>(xb, wqt, bq, 0.125f, qB, nullptr); // q/sqrt(64)
  gemm_bf16<<<gg, 256, 0, stream>>>(xb, wkt, bk, 1.0f,   kB, nullptr);
  gemm_bf16<<<gg, 256, 0, stream>>>(xb, wvt, bv, 1.0f,   vB, nullptr);

  attn_kernel<<<dim3(NHEAD, NBLK), 256, 0, stream>>>(qB, kB, vB, ctx);

  gemm_bf16<<<gg, 256, 0, stream>>>(ctx, wot, bo, 1.0f, nullptr, (float*)d_out);
}
